// DeformableConv2D_38147899523277
// MI455X (gfx1250) — matrix operand order repaired and checked by reading, not compiled, not run on hardware
//
#include <hip/hip_runtime.h>
#include <hip/hip_bf16.h>
#include <math.h>

typedef __attribute__((ext_vector_type(16))) __bf16 v16bf;
typedef __attribute__((ext_vector_type(8)))  __bf16 v8bf;
typedef __attribute__((ext_vector_type(4)))  __bf16 v4bf;
typedef __attribute__((ext_vector_type(8)))  float  v8f;
typedef __attribute__((ext_vector_type(4)))  float  v4f;
typedef __attribute__((ext_vector_type(4)))  unsigned int v4u;

namespace {
constexpr int Bn = 8, Hn = 128, Wn = 128, Cn = 64, Kn = 9, On = 64;
constexpr int CK      = Cn * Kn;      // 576
constexpr int KSTEPS  = CK / 32;      // 18 k-steps of 32
constexpr int TILE_H  = 4, TILE_W = 32;          // 128 pixels per workgroup
constexpr int PROWS   = TILE_H + 2, PCOLS = TILE_W + 2;
constexpr int PATCH_ELEMS = PROWS * PCOLS * Cn;  // 13056 bf16 = 26112 B LDS
constexpr int BOM_ELEMS   = KSTEPS * 2 * 32 * 16;   // 18432 (576 x 32 packed)
constexpr int BMAIN_ELEMS = KSTEPS * 4 * 32 * 16;   // 36864 (576 x 64 packed)
}

// ---------------------------------------------------------------------------
// Kernel 0: pack conv weights into WMMA B-fragment-native bf16 layout.
// B-frag (32x16 bf16, wave32): lane<16 -> N=lane, K {0-7,16-23};
//                              lane>=16 -> N=lane-16, K {8-15,24-31}.
// Row index r = tap*64 + channel  (tap = kh*3+kw).
// ---------------------------------------------------------------------------
__global__ void prep_weights(const float* __restrict__ offk,
                             const float* __restrict__ maskk,
                             const float* __restrict__ filt,
                             __bf16* __restrict__ bOM,
                             __bf16* __restrict__ bMain) {
  int idx = blockIdx.x * 256 + threadIdx.x;
  if (idx < BOM_ELEMS) {
    int t = idx & 15, lane = (idx >> 4) & 31, nt = (idx >> 9) & 1, kstep = idx >> 10;
    int n = nt * 16 + (lane & 15);
    int r = kstep * 32 + ((lane >> 4) & 1) * 8 + (t & 7) + (t >> 3) * 16;
    int k = r >> 6, c = r & 63;                 // tap, channel
    float v = 0.0f;
    if (n < 18)      v = offk[(k * 64 + c) * 18 + n];          // (KH,KW,C,18)
    else if (n < 27) v = maskk[(k * 64 + c) * 9 + (n - 18)];   // (KH,KW,C,9)
    bOM[idx] = (__bf16)v;
  } else {
    int j = idx - BOM_ELEMS;
    if (j >= BMAIN_ELEMS) return;
    int t = j & 15, lane = (j >> 4) & 31, nt = (j >> 9) & 3, kstep = j >> 11;
    int n = nt * 16 + (lane & 15);
    int r = kstep * 32 + ((lane >> 4) & 1) * 8 + (t & 7) + (t >> 3) * 16;
    int k = r >> 6, c = r & 63;
    int kh = k / 3, kw = k - kh * 3;
    bMain[j] = (__bf16)filt[((n * 64 + c) * 3 + kh) * 3 + kw]; // (OUT,C,KH,KW)
  }
}

// ---------------------------------------------------------------------------
// Kernel 1: offset+mask 3x3 conv as WMMA GEMM (M=pixels, N=32(27), K=576).
// A-frag (16x32 bf16): lane<16 -> M=lane, K {0-7,16-23}; lane>=16 -> K {8-15,24-31}.
// A from LDS halo patch, B staged in LDS, both on the ds pipe.
// ---------------------------------------------------------------------------
__global__ __launch_bounds__(256)
void dcn_offset_mask(const float* __restrict__ x,
                     const __bf16* __restrict__ bOM,
                     float* __restrict__ offws) {
  __shared__ __align__(16) __bf16 patch[PATCH_ELEMS];
  __shared__ __align__(16) __bf16 bsh[BOM_ELEMS];            // 36864 B
  int bid = blockIdx.x;
  int b = bid >> 7, tb = bid & 127;
  int tileY = tb >> 2, tileX = tb & 3;
  int tid = threadIdx.x;
  int y0 = tileY * TILE_H - 1, x0 = tileX * TILE_W - 1;

  // Stage packed B fragments into LDS (L2-resident source).
  {
    const v4u* src = (const v4u*)bOM;
    v4u* dst = (v4u*)bsh;
    for (int e = tid; e < BOM_ELEMS / 8; e += 256) dst[e] = src[e];
  }
  // Cooperative halo-patch load (6 x 34 x 64), SAME-pad with zero, cvt->bf16.
  for (int e = tid; e < PATCH_ELEMS; e += 256) {
    int c = e & 63, col = (e >> 6) % PCOLS, row = (e >> 6) / PCOLS;
    int gy = y0 + row, gx = x0 + col;
    float v = 0.0f;
    if (gy >= 0 && gy < Hn && gx >= 0 && gx < Wn)
      v = x[((b * Hn + gy) * Wn + gx) * Cn + c];
    patch[e] = (__bf16)v;
  }
  __syncthreads();

  int lane = tid & 31, wave = tid >> 5;
  int m = lane & 15, laneHi = lane >> 4;
  int pl = wave * 16 + m;                 // this lane's A-row pixel
  int ty = pl >> 5, tx = pl & 31;

  v8f c0 = {0.f,0.f,0.f,0.f,0.f,0.f,0.f,0.f};
  v8f c1 = {0.f,0.f,0.f,0.f,0.f,0.f,0.f,0.f};

  for (int kk = 0; kk < KSTEPS; ++kk) {
    int k = kk >> 1;                      // tap (each tap = 2 k-steps of 32 ch)
    int cb = (kk & 1) << 5;               // channel block base within tap
    int kh = k / 3, kw = k - kh * 3;
    int base = ((ty + kh) * PCOLS + (tx + kw)) * Cn + cb + (laneHi ? 8 : 0);
    v8bf r0 = *(const v8bf*)&patch[base];
    v8bf r1 = *(const v8bf*)&patch[base + 16];
    v16bf a;
#pragma unroll
    for (int t = 0; t < 8; ++t) { a[t] = r0[t]; a[t + 8] = r1[t]; }

    v16bf b0 = *(const v16bf*)(bsh + (size_t)((kk * 2 + 0) * 32 + lane) * 16);
    v16bf b1 = *(const v16bf*)(bsh + (size_t)((kk * 2 + 1) * 32 + lane) * 16);
    c0 = __builtin_amdgcn_wmma_f32_16x16x32_bf16(false, a, false, b0, (short)0, c0, false, false);
    c1 = __builtin_amdgcn_wmma_f32_16x16x32_bf16(false, a, false, b1, (short)0, c1, false, false);
  }

  // D-frag: VGPR i -> row M = i (+8 for lane>=16), col N = lane&15 (+16 for c1).
#pragma unroll
  for (int i = 0; i < 8; ++i) {
    int M = i + laneHi * 8;
    int p = wave * 16 + M;
    int gy = tileY * TILE_H + (p >> 5), gx = tileX * TILE_W + (p & 31);
    float* dst = offws + (size_t)((b * Hn + gy) * Wn + gx) * 32;
    dst[m] = c0[i];                                   // n = 0..15: dy/dx raw
    int n1 = 16 + m;
    float v = c1[i];
    if (n1 >= 18 && n1 < 27) v = 1.0f / (1.0f + __expf(-v));  // mask sigmoid
    dst[n1] = v;
  }
}

// ---------------------------------------------------------------------------
// Kernel 2: fused bilinear gather + main GEMM (M=pixels, N=64, K=576).
// Gather blend in packed (v4f -> v_pk_fma_f32) math; pk cvt to bf16;
// B fragments from LDS (ds pipe), gather corners from global (L2-resident).
// ---------------------------------------------------------------------------
__global__ __launch_bounds__(256)
void dcn_main(const float* __restrict__ x,
              const __bf16* __restrict__ bMain,
              const float* __restrict__ offws,
              float* __restrict__ out) {
  __shared__ __align__(16) __bf16 bsh[BMAIN_ELEMS];          // 73728 B
  int bid = blockIdx.x;
  int b = bid >> 7, tb = bid & 127;
  int tileY = tb >> 2, tileX = tb & 3;
  int tid = threadIdx.x;
  {
    const v4u* src = (const v4u*)bMain;
    v4u* dst = (v4u*)bsh;
    for (int e = tid; e < BMAIN_ELEMS / 8; e += 256) dst[e] = src[e];
  }
  __syncthreads();

  int lane = tid & 31, wave = tid >> 5;
  int m = lane & 15, laneHi = lane >> 4;
  int pl = wave * 16 + m;
  int ty = pl >> 5, tx = pl & 31;
  int gy = tileY * TILE_H + ty, gx = tileX * TILE_W + tx;
  const float* op = offws + (size_t)((b * Hn + gy) * Wn + gx) * 32;

  v8f acc0 = {0.f,0.f,0.f,0.f,0.f,0.f,0.f,0.f};
  v8f acc1 = acc0, acc2 = acc0, acc3 = acc0;

  for (int kk = 0; kk < KSTEPS; ++kk) {
    int k = kk >> 1;
    int cb = (kk & 1) << 5;
    int kh = k / 3, kw = k - kh * 3;

    float dy = op[2 * k], dx = op[2 * k + 1], mk = op[18 + k];
    float py = (float)(gy + kh - 1) + dy;
    float px = (float)(gx + kw - 1) + dx;
    float fy = floorf(py), fx = floorf(px);
    float ly = py - fy, lx = px - fx;
    int yi0 = (int)fy, xi0 = (int)fx, yi1 = yi0 + 1, xi1 = xi0 + 1;
    bool vy0 = (yi0 >= 0) && (yi0 < Hn), vy1 = (yi1 >= 0) && (yi1 < Hn);
    bool vx0 = (xi0 >= 0) && (xi0 < Wn), vx1 = (xi1 >= 0) && (xi1 < Wn);
    float w00 = (vy0 && vx0) ? (1.f - ly) * (1.f - lx) * mk : 0.f;
    float w01 = (vy0 && vx1) ? (1.f - ly) * lx * mk : 0.f;
    float w10 = (vy1 && vx0) ? ly * (1.f - lx) * mk : 0.f;
    float w11 = (vy1 && vx1) ? ly * lx * mk : 0.f;
    int yc0 = min(max(yi0, 0), Hn - 1), yc1 = min(max(yi1, 0), Hn - 1);
    int xc0 = min(max(xi0, 0), Wn - 1), xc1 = min(max(xi1, 0), Wn - 1);
    int ib = b * Hn;
    int b00 = ((ib + yc0) * Wn + xc0) * Cn, b01 = ((ib + yc0) * Wn + xc1) * Cn;
    int b10 = ((ib + yc1) * Wn + xc0) * Cn, b11 = ((ib + yc1) * Wn + xc1) * Cn;

    int s0 = cb + (laneHi ? 8 : 0);
    v16bf a;
#pragma unroll
    for (int run = 0; run < 2; ++run) {
      int c0 = s0 + run * 16;
      v4f q00a = *(const v4f*)(x + b00 + c0), q00b = *(const v4f*)(x + b00 + c0 + 4);
      v4f q01a = *(const v4f*)(x + b01 + c0), q01b = *(const v4f*)(x + b01 + c0 + 4);
      v4f q10a = *(const v4f*)(x + b10 + c0), q10b = *(const v4f*)(x + b10 + c0 + 4);
      v4f q11a = *(const v4f*)(x + b11 + c0), q11b = *(const v4f*)(x + b11 + c0 + 4);
      // Packed 4-corner blend: lowers to v_pk_fma_f32 pairs.
      v4f va = q00a * w00 + q01a * w01 + q10a * w10 + q11a * w11;
      v4f vb = q00b * w00 + q01b * w01 + q10b * w10 + q11b * w11;
      v4bf pa = __builtin_convertvector(va, v4bf);   // v_cvt_pk_bf16_f32
      v4bf pb = __builtin_convertvector(vb, v4bf);
#pragma unroll
      for (int j = 0; j < 4; ++j) {
        a[run * 8 + j]     = pa[j];
        a[run * 8 + 4 + j] = pb[j];
      }
    }

    const __bf16* bb = bsh + (size_t)(kk * 4 * 32 + lane) * 16;
    v16bf bf0 = *(const v16bf*)(bb);
    v16bf bf1 = *(const v16bf*)(bb + 512);
    v16bf bf2 = *(const v16bf*)(bb + 1024);
    v16bf bf3 = *(const v16bf*)(bb + 1536);
    acc0 = __builtin_amdgcn_wmma_f32_16x16x32_bf16(false, a, false, bf0, (short)0, acc0, false, false);
    acc1 = __builtin_amdgcn_wmma_f32_16x16x32_bf16(false, a, false, bf1, (short)0, acc1, false, false);
    acc2 = __builtin_amdgcn_wmma_f32_16x16x32_bf16(false, a, false, bf2, (short)0, acc2, false, false);
    acc3 = __builtin_amdgcn_wmma_f32_16x16x32_bf16(false, a, false, bf3, (short)0, acc3, false, false);
  }

  // NHWC epilogue: lane holds col o = nt*16 + m for 8 pixel-rows.
#pragma unroll
  for (int i = 0; i < 8; ++i) {
    int M = i + laneHi * 8;
    int p = wave * 16 + M;
    int gy2 = tileY * TILE_H + (p >> 5), gx2 = tileX * TILE_W + (p & 31);
    float* dst = out + (size_t)((b * Hn + gy2) * Wn + gx2) * On;
    dst[m]      = acc0[i];
    dst[16 + m] = acc1[i];
    dst[32 + m] = acc2[i];
    dst[48 + m] = acc3[i];
  }
}

extern "C" void kernel_launch(void* const* d_in, const int* in_sizes, int n_in,
                              void* d_out, int out_size, void* d_ws, size_t ws_size,
                              hipStream_t stream) {
  const float* x    = (const float*)d_in[0];
  const float* offk = (const float*)d_in[1];
  const float* mk   = (const float*)d_in[2];
  const float* filt = (const float*)d_in[3];
  float* out = (float*)d_out;

  __bf16* bOM   = (__bf16*)d_ws;
  __bf16* bMain = (__bf16*)((char*)d_ws + 2 * BOM_ELEMS);
  float*  offws = (float*)((char*)d_ws + 2 * BOM_ELEMS + 2 * BMAIN_ELEMS);

  int prepThreads = BOM_ELEMS + BMAIN_ELEMS;                  // 55296
  prep_weights<<<(prepThreads + 255) / 256, 256, 0, stream>>>(offk, mk, filt, bOM, bMain);

  int tiles = Bn * (Hn / TILE_H) * (Wn / TILE_W);             // 1024 workgroups
  dcn_offset_mask<<<tiles, 256, 0, stream>>>(x, bOM, offws);
  dcn_main<<<tiles, 256, 0, stream>>>(x, bMain, offws, out);
}
